// TransitionModel_77378130805138
// MI455X (gfx1250) — matrix operand order repaired and checked by reading, not compiled, not run on hardware
//
#include <hip/hip_runtime.h>
#include <cmath>

// ---- CDNA5 WMMA vector types -------------------------------------------------
typedef __attribute__((ext_vector_type(16))) __bf16       v16bf;
typedef __attribute__((ext_vector_type(8)))  float        v8f;
typedef __attribute__((ext_vector_type(4)))  unsigned int v4u;
typedef __attribute__((ext_vector_type(8)))  unsigned int v8u;

#define B_   256
#define T1_  128
#define S_   128
#define A_   32
#define H_   1024
#define E_   1024
#define MIN_STD 0.1f

__device__ __forceinline__ unsigned short f2bf(float f) {
  unsigned int u = __builtin_bit_cast(unsigned int, f);
  u += 0x7FFFu + ((u >> 16) & 1u);            // round-to-nearest-even
  return (unsigned short)(u >> 16);
}

__device__ __forceinline__ v16bf load_afrag(const unsigned short* p) {
  v4u alo = *(const v4u*)(p);
  v4u ahi = *(const v4u*)(p + 8);
  v8u r;
  r[0] = alo[0]; r[1] = alo[1]; r[2] = alo[2]; r[3] = alo[3];
  r[4] = ahi[0]; r[5] = ahi[1]; r[6] = ahi[2]; r[7] = ahi[3];
  return __builtin_bit_cast(v16bf, r);
}

// ---- one-time fp32 -> bf16 weight conversion --------------------------------
__global__ void convert_f32_bf16(const float* __restrict__ src,
                                 unsigned short* __restrict__ dst, int n) {
  int i = blockIdx.x * blockDim.x + threadIdx.x;
  if (i < n) dst[i] = f2bf(src[i]);
}

// ---- generic bf16 WMMA GEMM:  C = act(A[MxK] * W[NxK]^T + bias[N]) ----------
// Block = 4 waves (128 thr). Each wave owns a 32(M) x 64(N) strip:
// 8 accumulators, 2 A-fragments reused across 4 N-tiles, B-fragments reused
// across 2 M-tiles -> 8 WMMAs per (4 A + 8 B) b128 loads per K-chunk of 32.
// Fragment addressing: contiguous k order (lane half hi holds k = 16*hi + s) in both A and B.
__global__ __launch_bounds__(128)
void gemm_wmma(const unsigned short* __restrict__ Abf,  // M x K bf16, row-major
               const unsigned short* __restrict__ Wbf,  // N x K bf16, row-major
               const float* __restrict__ bias,          // N
               float* __restrict__ outF,                // M x N  (bf16out == 0)
               unsigned short* __restrict__ outB,       // M x N  (bf16out == 1)
               int M, int N, int K, int relu, int bf16out)
{
  const int lane = threadIdx.x & 31;
  const int wave = threadIdx.x >> 5;
  const int hi   = (lane >= 16) ? 1 : 0;
  const int lan  = lane & 15;
  const int m0   = blockIdx.y * 32;
  const int n0   = (blockIdx.x * 4 + wave) * 64;

  v8f c[2][4];
#pragma unroll
  for (int mi = 0; mi < 2; ++mi)
#pragma unroll
    for (int j = 0; j < 4; ++j) c[mi][j] = {};

  // A: lane<16 -> row m0+lan, K offsets {0..15}; lane>=16 -> {16..31}
  const unsigned short* aptr0 = Abf + (size_t)(m0 + lan) * K + hi * 16;
  const unsigned short* aptr1 = aptr0 + (size_t)16 * K;
  // B: column n = W row n; lanes<16 carry K k0..k0+15, lanes>=16 carry k0+16..k0+31
  const unsigned short* wptr = Wbf + (size_t)(n0 + lan) * K + hi * 16;
  const size_t wstride = (size_t)16 * K;     // next 16-wide N tile

  for (int k0 = 0; k0 < K; k0 += 32) {
    v16bf a0 = load_afrag(aptr0 + k0);
    v16bf a1 = load_afrag(aptr1 + k0);

#pragma unroll
    for (int j = 0; j < 4; ++j) {
      v8u braw = *(const v8u*)(wptr + (size_t)j * wstride + k0);
      v16bf b  = __builtin_bit_cast(v16bf, braw);
      c[0][j] = __builtin_amdgcn_wmma_f32_16x16x32_bf16(
                    false, a0, false, b, (short)0, c[0][j], false, false);
      c[1][j] = __builtin_amdgcn_wmma_f32_16x16x32_bf16(
                    false, a1, false, b, (short)0, c[1][j], false, false);
    }
  }

  // C/D layout: VGPR v -> M = m0 + mi*16 + v + 8*hi, N = n0 + j*16 + lan
#pragma unroll
  for (int j = 0; j < 4; ++j) {
    int n = n0 + j * 16 + lan;
    float bn = bias[n];
#pragma unroll
    for (int mi = 0; mi < 2; ++mi) {
#pragma unroll
      for (int v = 0; v < 8; ++v) {
        int m = m0 + mi * 16 + v + hi * 8;
        float val = c[mi][j][v] + bn;
        if (relu) val = val > 0.f ? val : 0.f;
        if (bf16out) outB[(size_t)m * N + n] = f2bf(val);
        else         outF[(size_t)m * N + n] = val;
      }
    }
  }
}

// ---- GRU cell elementwise (gate order r,z,n), plus [b1|o] concat staging ----
__global__ void gru_update(const float* __restrict__ GI, const float* __restrict__ GH,
                           float* __restrict__ Hf, unsigned short* __restrict__ Hbf,
                           unsigned short* __restrict__ HQcat,
                           const float* __restrict__ Obs, int t,
                           float* __restrict__ out_b1)
{
  int idx = blockIdx.x * blockDim.x + threadIdx.x;   // B_*H_ threads
  int b = idx >> 10;
  int j = idx & (H_ - 1);
  const float* gi = GI + (size_t)b * 3 * H_;
  const float* gh = GH + (size_t)b * 3 * H_;
  float r = 1.f / (1.f + __expf(-(gi[j] + gh[j])));
  float z = 1.f / (1.f + __expf(-(gi[H_ + j] + gh[H_ + j])));
  float n = tanhf(gi[2 * H_ + j] + r * gh[2 * H_ + j]);
  float b1 = (1.f - z) * n + z * Hf[idx];
  Hf[idx]  = b1;
  unsigned short b1b = f2bf(b1);
  Hbf[idx] = b1b;
  HQcat[(size_t)b * (H_ + E_) + j] = b1b;
  HQcat[(size_t)b * (H_ + E_) + H_ + j] =
      f2bf(Obs[((size_t)b * T1_ + t) * E_ + j]);
  out_b1[((size_t)b * T1_ + t) * H_ + j] = b1;
}

// ---- prior/posterior heads: split mu|std, softplus, reparam sample ----------
__global__ void dist_outputs(const float* __restrict__ PriRaw, const float* __restrict__ PosRaw,
                             const float* __restrict__ ep, const float* __restrict__ eq,
                             const float* __restrict__ Action, int t,
                             float* __restrict__ o_sp, float* __restrict__ o_mup,
                             float* __restrict__ o_stdp,
                             float* __restrict__ o_sq, float* __restrict__ o_muq,
                             float* __restrict__ o_stdq,
                             unsigned short* __restrict__ SA)
{
  int idx = blockIdx.x * blockDim.x + threadIdx.x;   // B_*S_ threads
  int b = idx >> 7;
  int j = idx & (S_ - 1);
  size_t obase = ((size_t)b * T1_ + t) * S_ + j;

  float mu_p = PriRaw[b * 2 * S_ + j];
  float h2   = PriRaw[b * 2 * S_ + S_ + j];
  float std_p = (h2 > 20.f ? h2 : log1pf(__expf(h2))) + MIN_STD;
  float sp = mu_p + std_p * ep[obase];
  o_mup[obase] = mu_p; o_stdp[obase] = std_p; o_sp[obase] = sp;

  float mu_q = PosRaw[b * 2 * S_ + j];
  float g2   = PosRaw[b * 2 * S_ + S_ + j];
  float std_q = (g2 > 20.f ? g2 : log1pf(__expf(g2))) + MIN_STD;
  float sq = mu_q + std_q * eq[obase];
  o_muq[obase] = mu_q; o_stdq[obase] = std_q; o_sq[obase] = sq;

  // feed next step: SA = [bf16(s_q) | bf16(a_{t+1})]
  SA[(size_t)b * (S_ + A_) + j] = f2bf(sq);
  if (j < A_) {
    float av = (t + 1 < T1_) ? Action[((size_t)b * T1_ + (t + 1)) * A_ + j] : 0.f;
    SA[(size_t)b * (S_ + A_) + S_ + j] = f2bf(av);
  }
}

// ---- t=0 state init ---------------------------------------------------------
__global__ void init_state(const float* __restrict__ s0, const float* __restrict__ b0,
                           const float* __restrict__ Action,
                           unsigned short* __restrict__ SA, float* __restrict__ Hf,
                           unsigned short* __restrict__ Hbf)
{
  int idx = blockIdx.x * blockDim.x + threadIdx.x;   // B_*H_ threads
  int b = idx >> 10;
  int j = idx & (H_ - 1);
  float h = b0[idx];
  Hf[idx] = h;
  Hbf[idx] = f2bf(h);
  if (j < S_) {
    SA[(size_t)b * (S_ + A_) + j] = f2bf(s0[(size_t)b * S_ + j]);
  } else if (j < S_ + A_) {
    SA[(size_t)b * (S_ + A_) + j] =
        f2bf(Action[((size_t)b * T1_ + 0) * A_ + (j - S_)]);
  }
}

// ---- driver -----------------------------------------------------------------
extern "C" void kernel_launch(void* const* d_in, const int* in_sizes, int n_in,
                              void* d_out, int out_size, void* d_ws, size_t ws_size,
                              hipStream_t stream)
{
  (void)in_sizes; (void)n_in; (void)out_size; (void)ws_size;
  const float* s0     = (const float*)d_in[0];
  const float* b0     = (const float*)d_in[1];
  const float* Act    = (const float*)d_in[2];
  const float* Obs    = (const float*)d_in[3];
  const float* ep     = (const float*)d_in[4];
  const float* eq     = (const float*)d_in[5];
  const float* W_sa   = (const float*)d_in[6];
  const float* b_sa   = (const float*)d_in[7];
  const float* W_ih   = (const float*)d_in[8];
  const float* W_hh   = (const float*)d_in[9];
  const float* b_ih   = (const float*)d_in[10];
  const float* b_hh   = (const float*)d_in[11];
  const float* W_bpri = (const float*)d_in[12];
  const float* b_bpri = (const float*)d_in[13];
  const float* W_spri = (const float*)d_in[14];
  const float* b_spri = (const float*)d_in[15];
  const float* W_bpos = (const float*)d_in[16];
  const float* b_bpos = (const float*)d_in[17];
  const float* W_spos = (const float*)d_in[18];
  const float* b_spos = (const float*)d_in[19];

  float* out = (float*)d_out;
  const size_t BTH = (size_t)B_ * T1_ * H_;
  const size_t BTS = (size_t)B_ * T1_ * S_;
  float* o_b1   = out;
  float* o_sp   = o_b1 + BTH;
  float* o_mup  = o_sp + BTS;
  float* o_stdp = o_mup + BTS;
  float* o_sq   = o_stdp + BTS;
  float* o_muq  = o_sq + BTS;
  float* o_stdq = o_muq + BTS;

  // workspace carve-out (256B aligned slots)
  char* ws = (char*)d_ws;
  size_t off = 0;
  auto take = [&](size_t bytes) -> void* {
    void* p = ws + off;
    off += (bytes + 255) & ~(size_t)255;
    return p;
  };
  unsigned short* Wsa_b   = (unsigned short*)take(2ull * H_ * (S_ + A_));
  unsigned short* Wih_b   = (unsigned short*)take(2ull * 3 * H_ * H_);
  unsigned short* Whh_b   = (unsigned short*)take(2ull * 3 * H_ * H_);
  unsigned short* Wbpri_b = (unsigned short*)take(2ull * H_ * H_);
  unsigned short* Wspri_b = (unsigned short*)take(2ull * 2 * S_ * H_);
  unsigned short* Wbpos_b = (unsigned short*)take(2ull * H_ * (H_ + E_));
  unsigned short* Wspos_b = (unsigned short*)take(2ull * 2 * S_ * H_);
  unsigned short* SA      = (unsigned short*)take(2ull * B_ * (S_ + A_));
  unsigned short* X       = (unsigned short*)take(2ull * B_ * H_);
  unsigned short* Hbf     = (unsigned short*)take(2ull * B_ * H_);
  unsigned short* HQcat   = (unsigned short*)take(2ull * B_ * (H_ + E_));
  unsigned short* HP      = (unsigned short*)take(2ull * B_ * H_);
  unsigned short* HQ      = (unsigned short*)take(2ull * B_ * H_);
  float* GI     = (float*)take(4ull * B_ * 3 * H_);
  float* GH     = (float*)take(4ull * B_ * 3 * H_);
  float* Hf     = (float*)take(4ull * B_ * H_);
  float* PriRaw = (float*)take(4ull * B_ * 2 * S_);
  float* PosRaw = (float*)take(4ull * B_ * 2 * S_);

  auto cvt = [&](const float* src, unsigned short* dst, size_t n) {
    convert_f32_bf16<<<dim3((unsigned)((n + 255) / 256)), dim3(256), 0, stream>>>(src, dst, (int)n);
  };
  // weights -> bf16, once per call; they then live in L2 (192MB) for all 128 steps
  cvt(W_sa,   Wsa_b,   (size_t)H_ * (S_ + A_));
  cvt(W_ih,   Wih_b,   (size_t)3 * H_ * H_);
  cvt(W_hh,   Whh_b,   (size_t)3 * H_ * H_);
  cvt(W_bpri, Wbpri_b, (size_t)H_ * H_);
  cvt(W_spri, Wspri_b, (size_t)2 * S_ * H_);
  cvt(W_bpos, Wbpos_b, (size_t)H_ * (H_ + E_));
  cvt(W_spos, Wspos_b, (size_t)2 * S_ * H_);

  init_state<<<dim3(B_ * H_ / 256), dim3(256), 0, stream>>>(s0, b0, Act, SA, Hf, Hbf);

  const dim3 blk(128);
  const dim3 gN1024(H_ / 256, B_ / 32);      // N=1024
  const dim3 gN3072(3 * H_ / 256, B_ / 32);  // N=3072
  const dim3 gN256(2 * S_ / 256, B_ / 32);   // N=256

  for (int t = 0; t < T1_; ++t) {
    // x = relu([s|a] Wsa^T + b)                 (256 x 160 x 1024)
    gemm_wmma<<<gN1024, blk, 0, stream>>>(SA, Wsa_b, b_sa, nullptr, X,
                                          B_, H_, S_ + A_, 1, 1);
    // gi = x Wih^T + b_ih ; gh = h Whh^T + b_hh (256 x 1024 x 3072)
    gemm_wmma<<<gN3072, blk, 0, stream>>>(X, Wih_b, b_ih, GI, nullptr,
                                          B_, 3 * H_, H_, 0, 0);
    gemm_wmma<<<gN3072, blk, 0, stream>>>(Hbf, Whh_b, b_hh, GH, nullptr,
                                          B_, 3 * H_, H_, 0, 0);
    gru_update<<<dim3(B_ * H_ / 256), dim3(256), 0, stream>>>(GI, GH, Hf, Hbf,
                                                              HQcat, Obs, t, o_b1);
    // prior head
    gemm_wmma<<<gN1024, blk, 0, stream>>>(Hbf, Wbpri_b, b_bpri, nullptr, HP,
                                          B_, H_, H_, 1, 1);
    gemm_wmma<<<gN256, blk, 0, stream>>>(HP, Wspri_b, b_spri, PriRaw, nullptr,
                                         B_, 2 * S_, H_, 0, 0);
    // posterior head ([b1|o] concat staged in gru_update)
    gemm_wmma<<<gN1024, blk, 0, stream>>>(HQcat, Wbpos_b, b_bpos, nullptr, HQ,
                                          B_, H_, H_ + E_, 1, 1);
    gemm_wmma<<<gN256, blk, 0, stream>>>(HQ, Wspos_b, b_spos, PosRaw, nullptr,
                                         B_, 2 * S_, H_, 0, 0);
    dist_outputs<<<dim3(B_ * S_ / 256), dim3(256), 0, stream>>>(
        PriRaw, PosRaw, ep, eq, Act, t,
        o_sp, o_mup, o_stdp, o_sq, o_muq, o_stdq, SA);
  }
}
